// SAE_loss_CL_8546984919260
// MI455X (gfx1250) — compile-verified, matrix operand order checked by reading, not run on hardware
//
#include <hip/hip_runtime.h>
#include <math.h>

typedef int v8i __attribute__((ext_vector_type(8)));

#define BATCH 2048
#define DCOL  256
#define DTCOL 768

// ---------------- reductions (wave32) ----------------
__device__ __forceinline__ float waveSum(float v) {
#pragma unroll
  for (int off = 16; off > 0; off >>= 1) v += __shfl_down(v, off, 32);
  return v;
}
__device__ __forceinline__ float waveMax(float v) {
#pragma unroll
  for (int off = 16; off > 0; off >>= 1) v = fmaxf(v, __shfl_down(v, off, 32));
  return v;
}
__device__ __forceinline__ float waveMin(float v) {
#pragma unroll
  for (int off = 16; off > 0; off >>= 1) v = fminf(v, __shfl_down(v, off, 32));
  return v;
}
__device__ __forceinline__ float blockSum(float v, float* buf) {
  int lane = threadIdx.x & 31, wv = threadIdx.x >> 5;
  v = waveSum(v);
  if (lane == 0) buf[wv] = v;
  __syncthreads();
  float r = (threadIdx.x < 8) ? buf[threadIdx.x] : 0.0f;
  if (wv == 0) r = waveSum(r);
  if (threadIdx.x == 0) buf[0] = r;
  __syncthreads();
  r = buf[0];
  __syncthreads();
  return r;
}
__device__ __forceinline__ float blockMax(float v, float* buf) {
  int lane = threadIdx.x & 31, wv = threadIdx.x >> 5;
  v = waveMax(v);
  if (lane == 0) buf[wv] = v;
  __syncthreads();
  float r = (threadIdx.x < 8) ? buf[threadIdx.x] : -3.4e38f;
  if (wv == 0) r = waveMax(r);
  if (threadIdx.x == 0) buf[0] = r;
  __syncthreads();
  r = buf[0];
  __syncthreads();
  return r;
}
__device__ __forceinline__ float blockMin(float v, float* buf) {
  int lane = threadIdx.x & 31, wv = threadIdx.x >> 5;
  v = waveMin(v);
  if (lane == 0) buf[wv] = v;
  __syncthreads();
  float r = (threadIdx.x < 8) ? buf[threadIdx.x] : 3.4e38f;
  if (wv == 0) r = waveMin(r);
  if (threadIdx.x == 0) buf[0] = r;
  __syncthreads();
  r = buf[0];
  __syncthreads();
  return r;
}

// ---------------- 1) per-block partial mins of data1/2/3 ----------------
__global__ __launch_bounds__(256) void min_part_kernel(
    const float* __restrict__ a, const float* __restrict__ b,
    const float* __restrict__ c, float* __restrict__ part) {
  const float* x = (blockIdx.y == 0) ? a : (blockIdx.y == 1) ? b : c;
  const int n = BATCH * DCOL;
  float m = 3.4e38f;
  for (int i = blockIdx.x * blockDim.x + threadIdx.x; i < n;
       i += gridDim.x * blockDim.x)
    m = fminf(m, x[i]);
  __shared__ float sh[8];
  m = blockMin(m, sh);
  if (threadIdx.x == 0) part[blockIdx.y * 512 + blockIdx.x] = m;
}

// ---------------- 2) final mins ----------------
__global__ __launch_bounds__(256) void min_final_kernel(
    const float* __restrict__ part, float* __restrict__ mins) {
  __shared__ float sh[8];
  int t = threadIdx.x;
  for (int s = 0; s < 3; s++) {
    float m = 3.4e38f;
    for (int i = t; i < 512; i += 256) m = fminf(m, part[s * 512 + i]);
    m = blockMin(m, sh);
    if (t == 0) mins[s] = m;
    __syncthreads();
  }
}

// ---------------- 3) joint-entropy counts via V_WMMA_I32_16X16X64_IU8 ----
// One block = one batch row. counts_S = eq_S @ ones  (IU8 WMMA, B = ones).
// Only the 3 base equality masks are packed from compares; the 4 composite
// sets are derived with bytewise AND of the packed words (bytes are 0/1).
__global__ __launch_bounds__(256) void entropy_kernel(
    const float* __restrict__ d1, const float* __restrict__ d2,
    const float* __restrict__ d3, const float* __restrict__ mins,
    float* __restrict__ rowH /* [7][BATCH] */) {
  __shared__ int4 sbins[DCOL];
  __shared__ float sred[8 * 7];
  const int row = blockIdx.x;
  const int t = threadIdx.x;

  const float l1 = floorf(mins[0]);
  const float l2 = floorf(mins[1]);
  const float l3 = floorf(mins[2]);
  const float step = 0.01f;
  int b1 = (int)floorf((d1[row * DCOL + t] - l1) / step);
  int b2 = (int)floorf((d2[row * DCOL + t] - l2) / step);
  int b3 = (int)floorf((d3[row * DCOL + t] - l3) / step);
  sbins[t] = make_int4(b1, b2, b3, 0);
  __syncthreads();

  const int lane = t & 31;
  const int wv = t >> 5;   // 0..7 waves
  const int m = lane & 15; // A-matrix M row of this lane
  const int h = lane >> 4; // lane half -> which 32 j's this lane supplies

  v8i ones;
#pragma unroll
  for (int e = 0; e < 8; e++) ones[e] = 0x01010101;

  float part[7];
#pragma unroll
  for (int s = 0; s < 7; s++) part[s] = 0.0f;

#pragma unroll 1
  for (int mt = wv; mt < 16; mt += 8) {  // 16 M-tiles of 16 i's
    const int i = mt * 16 + m;
    const int4 bi = sbins[i];
    v8i acc[7];
#pragma unroll
    for (int s = 0; s < 7; s++)
#pragma unroll
      for (int e = 0; e < 8; e++) acc[s][e] = 0;

#pragma unroll
    for (int ks = 0; ks < 4; ks++) {  // K = 64 j's per step
      const int jbase = ks * 64 + h * 32;
      v8i A0, A1, A2;
#pragma unroll
      for (int v = 0; v < 8; v++) {
        unsigned w0 = 0, w1 = 0, w2 = 0;
#pragma unroll
        for (int tb = 0; tb < 4; tb++) {
          const int j = jbase + v * 4 + tb;
          const int4 bj = sbins[j];
          const int sh = 8 * tb;
          w0 |= (bi.x == bj.x ? 1u : 0u) << sh;
          w1 |= (bi.y == bj.y ? 1u : 0u) << sh;
          w2 |= (bi.z == bj.z ? 1u : 0u) << sh;
        }
        A0[v] = (int)w0;
        A1[v] = (int)w1;
        A2[v] = (int)w2;
      }
      acc[0] = __builtin_amdgcn_wmma_i32_16x16x64_iu8(false, A0, false, ones,
                                                      acc[0], false, false);
      acc[1] = __builtin_amdgcn_wmma_i32_16x16x64_iu8(false, A1, false, ones,
                                                      acc[1], false, false);
      acc[2] = __builtin_amdgcn_wmma_i32_16x16x64_iu8(false, A2, false, ones,
                                                      acc[2], false, false);
      v8i T = A0 & A2;  // {1,3}
      acc[3] = __builtin_amdgcn_wmma_i32_16x16x64_iu8(false, T, false, ones,
                                                      acc[3], false, false);
      T = A1 & A2;      // {2,3}
      acc[4] = __builtin_amdgcn_wmma_i32_16x16x64_iu8(false, T, false, ones,
                                                      acc[4], false, false);
      T = A0 & A1;      // {1,2}
      acc[5] = __builtin_amdgcn_wmma_i32_16x16x64_iu8(false, T, false, ones,
                                                      acc[5], false, false);
      T = T & A2;       // {1,2,3}
      acc[6] = __builtin_amdgcn_wmma_i32_16x16x64_iu8(false, T, false, ones,
                                                      acc[6], false, false);
    }
    // D layout: VGPR r holds counts for M=r+8h (columns duplicated 16x).
#pragma unroll
    for (int s = 0; s < 7; s++) {
      float ls = 0.0f;
#pragma unroll
      for (int r = 0; r < 8; r++)
        ls += __logf((float)acc[s][r] * (1.0f / 256.0f));
      part[s] += ls;
    }
  }

#pragma unroll
  for (int s = 0; s < 7; s++) {
    float v = waveSum(part[s]);
    if (lane == 0) sred[wv * 7 + s] = v;
  }
  __syncthreads();
  if (t < 7) {
    float tot = 0.0f;
    for (int w = 0; w < 8; w++) tot += sred[w * 7 + t];
    // tot = 16 * sum_i log(counts_i/256);  H = -(1/256) sum_i log(...)
    rowH[t * BATCH + row] = -tot * (1.0f / 4096.0f);
  }
}

// ---------------- 4) _h_soft terms (7 pairs, incl. concatenations) -------
__global__ __launch_bounds__(256) void hsoft_kernel(
    const float* __restrict__ d1, const float* __restrict__ d2,
    const float* __restrict__ d3, const float* __restrict__ o1,
    const float* __restrict__ o2, const float* __restrict__ o3,
    float* __restrict__ rowh /* [7][BATCH] */) {
  __shared__ float sh[8];
  const int row = blockIdx.x, p = blockIdx.y, t = threadIdx.x;
  const float* ds[3];
  const float* os[3];
  int nseg;
  switch (p) {
    case 0: ds[0] = d1; os[0] = o1; nseg = 1; break;
    case 1: ds[0] = d2; os[0] = o2; nseg = 1; break;
    case 2: ds[0] = d3; os[0] = o3; nseg = 1; break;
    case 3: ds[0] = d1; ds[1] = d3; os[0] = o1; os[1] = o3; nseg = 2; break;
    case 4: ds[0] = d2; ds[1] = d3; os[0] = o2; os[1] = o3; nseg = 2; break;
    case 5: ds[0] = d1; ds[1] = d2; os[0] = o1; os[1] = o2; nseg = 2; break;
    default:
      ds[0] = d1; ds[1] = d2; ds[2] = d3;
      os[0] = o1; os[1] = o2; os[2] = o3; nseg = 3; break;
  }
  const int C = nseg * DCOL;

  float md = -3.4e38f, mo = -3.4e38f;
  for (int c = t; c < C; c += 256) {
    const int sg = c >> 8, off = c & 255;
    md = fmaxf(md, ds[sg][row * DCOL + off]);
    mo = fmaxf(mo, os[sg][row * DCOL + off]);
  }
  md = blockMax(md, sh);
  mo = blockMax(mo, sh);

  float sd = 0.f, so = 0.f, u1 = 0.f, u2 = 0.f;
  for (int c = t; c < C; c += 256) {
    const int sg = c >> 8, off = c & 255;
    const float dd = ds[sg][row * DCOL + off] - md;
    const float oo = os[sg][row * DCOL + off] - mo;
    const float ed = __expf(dd);
    sd += ed;
    so += __expf(oo);
    u1 += ed * oo;
    u2 += ed * dd;
  }
  sd = blockSum(sd, sh);
  so = blockSum(so, sh);
  u1 = blockSum(u1, sh);
  u2 = blockSum(u2, sh);

  if (t == 0) {
    const float t1 = u1 / sd - __logf(so);  // sum_c p*logq
    const float t2 = u2 / sd - __logf(sd);  // sum_c p*logp
    const float invC = 1.0f / (float)C;
    rowh[p * BATCH + row] = -(1.0f - invC) * t1 - invC * t2;
  }
}

// ---------------- 5) MSE partials ----------------
__global__ __launch_bounds__(256) void mse_kernel(
    const float* __restrict__ data, const float* __restrict__ outp,
    float* __restrict__ part) {
  const int n = BATCH * DTCOL;
  float s = 0.0f;
  for (int i = blockIdx.x * blockDim.x + threadIdx.x; i < n;
       i += gridDim.x * blockDim.x) {
    const float df = data[i] - outp[i];
    s += df * df;
  }
  __shared__ float sh[8];
  s = blockSum(s, sh);
  if (threadIdx.x == 0) part[blockIdx.x] = s;
}

// ---------------- 6) final combine ----------------
__global__ __launch_bounds__(256) void final_kernel(
    const float* __restrict__ entRows, const float* __restrict__ hRows,
    const float* __restrict__ msePart, float* __restrict__ out) {
  __shared__ float sh[8];
  __shared__ float Hsum[7], hmean[7];
  const int t = threadIdx.x;
  for (int s = 0; s < 7; s++) {
    float v = 0.f;
    for (int i = t; i < BATCH; i += 256) v += entRows[s * BATCH + i];
    v = blockSum(v, sh);
    if (t == 0) Hsum[s] = v;
    float w = 0.f;
    for (int i = t; i < BATCH; i += 256) w += hRows[s * BATCH + i];
    w = blockSum(w, sh);
    if (t == 0) hmean[s] = w / (float)BATCH;
  }
  float ms = 0.f;
  for (int i = t; i < 512; i += 256) ms += msePart[i];
  ms = blockSum(ms, sh);

  if (t == 0) {
    float Hb[7];
    for (int s = 0; s < 7; s++) Hb[s] = Hsum[s] / (float)BATCH;
    const float mse_loss = 0.5f * ms / (float)(BATCH * DTCOL);
    const float H1 = Hb[0] - hmean[0];
    const float H2 = Hb[1] - hmean[1];
    const float H3 = Hb[2] - hmean[2];
    const float MI13 = (hmean[0] + hmean[2] - hmean[3]) - (Hb[0] + Hb[2] - Hb[3]);
    const float MI23 = (hmean[1] + hmean[2] - hmean[4]) - (Hb[1] + Hb[2] - Hb[4]);
    const float MI12 = (hmean[0] + hmean[1] - hmean[5]) - (Hb[0] + Hb[1] - Hb[5]);
    const float aveD = Hsum[3] + Hsum[4] - Hsum[2] - Hsum[6];  // row sums
    const float aveL = hmean[4] - hmean[2] + hmean[3] - hmean[6];
    const float CMI = aveL - aveD;
    out[0] = 0.5f * mse_loss + 0.25f * (H1 * H1 + H2 * H2 + H3 * H3) +
             0.25f * (MI13 * MI13 + MI23 * MI23 + MI12 * MI12 + CMI * CMI);
  }
}

extern "C" void kernel_launch(void* const* d_in, const int* in_sizes, int n_in,
                              void* d_out, int out_size, void* d_ws,
                              size_t ws_size, hipStream_t stream) {
  const float* data = (const float*)d_in[0];  // [2048,768]
  const float* d1 = (const float*)d_in[1];    // [2048,256]
  const float* d2 = (const float*)d_in[2];
  const float* d3 = (const float*)d_in[3];
  const float* o1 = (const float*)d_in[4];
  const float* o2 = (const float*)d_in[5];
  const float* o3 = (const float*)d_in[6];
  const float* outp = (const float*)d_in[7];  // [2048,768]

  float* ws = (float*)d_ws;
  float* mins = ws;                 // 3
  float* minpart = ws + 16;         // 3*512
  float* entRows = ws + 2048;       // 7*2048
  float* hRows = ws + 16384;        // 7*2048
  float* msePart = ws + 30720;      // 512

  dim3 gMin(512, 3);
  min_part_kernel<<<gMin, 256, 0, stream>>>(d1, d2, d3, minpart);
  min_final_kernel<<<1, 256, 0, stream>>>(minpart, mins);
  entropy_kernel<<<BATCH, 256, 0, stream>>>(d1, d2, d3, mins, entRows);
  dim3 gH(BATCH, 7);
  hsoft_kernel<<<gH, 256, 0, stream>>>(d1, d2, d3, o1, o2, o3, hRows);
  mse_kernel<<<512, 256, 0, stream>>>(data, outp, msePart);
  final_kernel<<<1, 256, 0, stream>>>(entRows, hRows, msePart, (float*)d_out);
}
